// Transformer_13726715478448
// MI455X (gfx1250) — matrix operand order repaired and checked by reading, not compiled, not run on hardware
//
#include <hip/hip_runtime.h>
#include <hip/hip_bf16.h>

// ---------------------------------------------------------------------------
// Transformer forward for MI455X (gfx1250, wave32, WMMA).
// All GEMMs route through one bf16-WMMA kernel (v_wmma_f32_16x16x32_bf16)
// with a register-staged, double-buffered LDS pipeline (1 barrier / K-step).
// ---------------------------------------------------------------------------

typedef __attribute__((ext_vector_type(16))) __bf16 v16bf;
typedef __attribute__((ext_vector_type(8)))  float  v8f;

#define TBM 128
#define TBN 128
#define TBK 32
#define LDT 34          // padded LDS row stride (bf16 elems) -> conflict-free frag reads

__device__ __forceinline__ unsigned short f2bf(float f) {
  unsigned int u = __float_as_uint(f);
  unsigned int r = u + 0x7fffu + ((u >> 16) & 1u);   // round-to-nearest-even
  return (unsigned short)(r >> 16);
}
__device__ __forceinline__ unsigned int pack2bf(float lo, float hi) {
  return (unsigned int)f2bf(lo) | ((unsigned int)f2bf(hi) << 16);
}

// C[z] = act(A[z] (MxK, f32) x B[z] + bias), B f32 is KxN (transB=0) or NxK (transB=1).
// Operands converted to bf16 while staging into LDS; f32 accumulation via WMMA.
// Pipeline per K-step: barrier -> issue next-slab global loads (regs) ->
//                      8 WMMAs on current LDS buffer -> convert+store next buffer.
__global__ __launch_bounds__(256)
void gemm_bf16_wmma(const float* __restrict__ Ag, const float* __restrict__ Bg,
                    const float* __restrict__ bias, float* __restrict__ Cg,
                    int M, int N, int K,
                    long long sA, long long sB, long long sC,
                    int transB, int relu)
{
  __shared__ __bf16 As[2][TBM * LDT];   // [m][k]
  __shared__ __bf16 Bs[2][TBN * LDT];   // [n][k]  (K-contiguous for both operands)

  const float* A = Ag + (long long)blockIdx.z * sA;
  const float* B = Bg + (long long)blockIdx.z * sB;
  float*       C = Cg + (long long)blockIdx.z * sC;

  const int bm   = blockIdx.y * TBM;
  const int bn   = blockIdx.x * TBN;
  const int tid  = threadIdx.x;
  const int lane = tid & 31;
  const int w    = tid >> 5;          // 8 wave32 waves
  const int wm   = (w & 3) * 32;      // wave tile: 32 (M) x 64 (N)
  const int wn   = (w >> 2) * 64;

  v8f acc[2][4];
#pragma unroll
  for (int i = 0; i < 2; ++i)
#pragma unroll
    for (int j = 0; j < 4; ++j)
      acc[i][j] = (v8f){0.f,0.f,0.f,0.f,0.f,0.f,0.f,0.f};

  // loader index maps (256 threads, 16 f32 each)
  const int ar  = tid >> 1;           // A: row, half-K
  const int ahk = (tid & 1) * 16;
  const int bkg = (tid >> 5) * 4;     // B KxN: wave w owns k-rows 4w..4w+3
  const int bng = (tid & 31) * 4;     //        lane owns 4 consecutive n
  const int brr = tid >> 1;           // B NxK: n-row, half-K
  const int bhk = (tid & 1) * 16;

  float ra[16], rb[16];               // register staging for next K-slab

  auto load_tiles = [&](int k0) {
    {
      const float4* p4 = (const float4*)(A + (long long)(bm + ar) * K + (k0 + ahk));
#pragma unroll
      for (int j = 0; j < 4; ++j) {
        float4 t = p4[j];
        ra[4*j] = t.x; ra[4*j+1] = t.y; ra[4*j+2] = t.z; ra[4*j+3] = t.w;
      }
      if (k0 + 2 * TBK < K) __builtin_prefetch((const float*)p4 + 2 * TBK, 0, 1);
    }
    if (!transB) {
      if (bn + bng + 3 < N) {         // full tile: wave reads 4 contiguous 512B rows
#pragma unroll
        for (int r = 0; r < 4; ++r) {
          float4 t = *(const float4*)(B + (long long)(k0 + bkg + r) * N + (bn + bng));
          rb[4*r] = t.x; rb[4*r+1] = t.y; rb[4*r+2] = t.z; rb[4*r+3] = t.w;
        }
        if (k0 + 2 * TBK < K)
          __builtin_prefetch(B + (long long)(k0 + 2 * TBK + bkg) * N + bn + bng, 0, 1);
      } else {                        // edge (e.g. N=64 for P·V): clamp reads
#pragma unroll
        for (int r = 0; r < 4; ++r)
#pragma unroll
          for (int e = 0; e < 4; ++e) {
            int nc = bn + bng + e; if (nc >= N) nc = N - 1;
            rb[4*r + e] = B[(long long)(k0 + bkg + r) * N + nc];
          }
      }
    } else {                          // B stored NxK (K-matrix for Q·K^T)
      int nn = bn + brr; if (nn >= N) nn = N - 1;
      const float4* p4 = (const float4*)(B + (long long)nn * K + (k0 + bhk));
#pragma unroll
      for (int j = 0; j < 4; ++j) {
        float4 t = p4[j];
        rb[4*j] = t.x; rb[4*j+1] = t.y; rb[4*j+2] = t.z; rb[4*j+3] = t.w;
      }
      if (k0 + 2 * TBK < K) __builtin_prefetch((const float*)p4 + 2 * TBK, 0, 1);
    }
  };

  auto store_tiles = [&](int buf) {
    unsigned int* AsU = (unsigned int*)As[buf];
#pragma unroll
    for (int j = 0; j < 4; ++j) {
      int base = ar * LDT + ahk + 4 * j;
      AsU[base >> 1]       = pack2bf(ra[4*j],   ra[4*j+1]);
      AsU[(base + 2) >> 1] = pack2bf(ra[4*j+2], ra[4*j+3]);
    }
    unsigned int* BsU = (unsigned int*)Bs[buf];
    if (!transB) {                    // in-register 4x4 transpose -> Bs[n][k]
#pragma unroll
      for (int e = 0; e < 4; ++e) {
        int base = (bng + e) * LDT + bkg;
        BsU[base >> 1]       = pack2bf(rb[e],     rb[4 + e]);
        BsU[(base + 2) >> 1] = pack2bf(rb[8 + e], rb[12 + e]);
      }
    } else {
#pragma unroll
      for (int j = 0; j < 4; ++j) {
        int base = brr * LDT + bhk + 4 * j;
        BsU[base >> 1]       = pack2bf(rb[4*j],   rb[4*j+1]);
        BsU[(base + 2) >> 1] = pack2bf(rb[4*j+2], rb[4*j+3]);
      }
    }
  };

  const int lm  = lane & 15;
  const int hiA = (lane >> 4) * 8;    // A frag: lanes 16-31 hold K+8 / K+24
  const int hiB = (lane >> 4) * 16;   // B frag: same K order as A (lanes 16-31 hold K+8 / K+24)

  auto compute = [&](int buf) {       // ISA 7.12.2 16-bit A/B layouts, 8 WMMAs
    v16bf afr[2], bfr[4];
#pragma unroll
    for (int tm = 0; tm < 2; ++tm) {
      unsigned int* au = (unsigned int*)&afr[tm];
      const unsigned int* rowp = (const unsigned int*)(As[buf] + (wm + tm * 16 + lm) * LDT);
#pragma unroll
      for (int p = 0; p < 8; ++p) {
        int kk = ((p >= 4) ? 16 : 0) + hiA + (p & 3) * 2;
        au[p] = rowp[kk >> 1];
      }
    }
#pragma unroll
    for (int tn = 0; tn < 4; ++tn) {
      unsigned int* bu = (unsigned int*)&bfr[tn];
      const unsigned int* rowp = (const unsigned int*)(Bs[buf] + (wn + tn * 16 + lm) * LDT);
#pragma unroll
      for (int p = 0; p < 8; ++p) bu[p] = rowp[(((p >= 4) ? 16 : 0) + hiA + (p & 3) * 2) >> 1];
    }
#pragma unroll
    for (int tm = 0; tm < 2; ++tm)
#pragma unroll
      for (int tn = 0; tn < 4; ++tn)
        acc[tm][tn] = __builtin_amdgcn_wmma_f32_16x16x32_bf16(
            false, afr[tm], false, bfr[tn], (short)0, acc[tm][tn], false, false);
  };

  // software pipeline: prologue stage, then 1 barrier per K-step
  const int nIter = K / TBK;          // all K in this model are multiples of 32
  load_tiles(0);
  store_tiles(0);
#pragma unroll 1
  for (int it = 0; it < nIter; ++it) {
    __syncthreads();                  // buf(it) visible; buf(it^1) free to overwrite
    const int buf = it & 1;
    if (it + 1 < nIter) load_tiles((it + 1) * TBK);   // global -> regs (async)
    compute(buf);                                     // WMMAs hide load latency
    if (it + 1 < nIter) store_tiles(buf ^ 1);         // regs -> other LDS buffer
  }

  // ---- epilogue: C layout (lanes 0-15: M=vgpr; 16-31: M=vgpr+8), bias + relu
  const int moff = (lane >> 4) * 8;
#pragma unroll
  for (int tm = 0; tm < 2; ++tm) {
#pragma unroll
    for (int tn = 0; tn < 4; ++tn) {
      int gn = bn + wn + tn * 16 + lm;
      if (gn < N) {
        float bv = bias ? bias[gn] : 0.f;
#pragma unroll
        for (int e = 0; e < 8; ++e) {
          int gm = bm + wm + tm * 16 + moff + e;
          float val = acc[tm][tn][e] + bv;
          if (relu) val = fmaxf(val, 0.f);
          C[(long long)gm * N + gn] = val;
        }
      }
    }
  }
}

// --------------------------- helper kernels --------------------------------

__device__ __forceinline__ float block_sum(float v, float* sm) {
  int t = threadIdx.x; sm[t] = v; __syncthreads();
  for (int s = 128; s > 0; s >>= 1) { if (t < s) sm[t] += sm[t + s]; __syncthreads(); }
  float r = sm[0]; __syncthreads(); return r;
}
__device__ __forceinline__ float block_max(float v, float* sm) {
  int t = threadIdx.x; sm[t] = v; __syncthreads();
  for (int s = 128; s > 0; s >>= 1) { if (t < s) sm[t] = fmaxf(sm[t], sm[t + s]); __syncthreads(); }
  float r = sm[0]; __syncthreads(); return r;
}

__global__ __launch_bounds__(256)
void embed_kernel(const int* __restrict__ tok, const float* __restrict__ emb,
                  float* __restrict__ out, int Dm) {
  int t = blockIdx.x;
  long long row = tok[t];
  for (int d = threadIdx.x; d < Dm; d += 256)
    out[(long long)t * Dm + d] = emb[row * Dm + d];
}

// in-place RoPE on [B,S,H,DK] (== [B,S,D]); pos = token % S
__global__ __launch_bounds__(256)
void rope_kernel(float* __restrict__ x, long long npairs, int S, int Hn, int DKn, int Dm) {
  long long i = (long long)blockIdx.x * 256 + threadIdx.x;
  if (i >= npairs) return;
  int halfdk = DKn >> 1;
  int p = (int)(i % halfdk);
  long long r = i / halfdk;
  int h = (int)(r % Hn);
  long long t = r / Hn;
  int pos = (int)(t % S);
  float inv = __powf(10000.0f, -(2.0f * (float)p) / (float)DKn);
  float ang = (float)pos * inv;
  float c, sn; __sincosf(ang, &sn, &c);
  float* b = x + t * Dm + h * DKn + 2 * p;
  float x1 = b[0], x2 = b[1];
  b[0] = x1 * c - x2 * sn;
  b[1] = x1 * sn + x2 * c;
}

__global__ __launch_bounds__(256)
void to_heads_kernel(const float* __restrict__ in, float* __restrict__ out,
                     long long total, int S, int Hn, int DKn) {
  long long i = (long long)blockIdx.x * 256 + threadIdx.x;
  if (i >= total) return;
  int d = (int)(i % DKn); long long r = i / DKn;
  int h = (int)(r % Hn);  long long r2 = r / Hn;
  int s = (int)(r2 % S);  long long b = r2 / S;
  out[(((long long)b * Hn + h) * S + s) * DKn + d] = in[i];
}

__global__ __launch_bounds__(256)
void from_heads_kernel(const float* __restrict__ in, float* __restrict__ out,
                       long long total, int S, int Hn, int DKn) {
  long long i = (long long)blockIdx.x * 256 + threadIdx.x;
  if (i >= total) return;
  int d = (int)(i % DKn); long long r = i / DKn;
  int s = (int)(r % S);   long long r2 = r / S;
  int h = (int)(r2 % Hn); long long b = r2 / Hn;
  out[(((long long)b * S + s) * Hn + h) * DKn + d] = in[i];
}

// one block per score row (b,h,q): scale, pad(+causal) mask, softmax over k
__global__ __launch_bounds__(256)
void attn_softmax_kernel(float* __restrict__ sc, const int* __restrict__ toks,
                         int Sq, int Sk, int Hn, int causal, float scale) {
  __shared__ float sm[256];
  long long row = blockIdx.x;
  int q = (int)(row % Sq);
  int b = (int)(row / ((long long)Sq * Hn));
  float* r = sc + row * (long long)Sk;
  const int* tk = toks + (long long)b * Sk;
  float lv[4]; int cnt = 0; float m = -3.4e38f;
  for (int k = threadIdx.x; k < Sk; k += 256) {
    float v = r[k] * scale;
    bool ok = (tk[k] != 0) && (!causal || (k <= q));
    v = ok ? v : -1.0e9f;
    lv[cnt++] = v; m = fmaxf(m, v);
  }
  m = block_max(m, sm);
  float s = 0.f;
  for (int i = 0; i < cnt; ++i) { lv[i] = __expf(lv[i] - m); s += lv[i]; }
  s = block_sum(s, sm);
  float inv = 1.f / s;
  cnt = 0;
  for (int k = threadIdx.x; k < Sk; k += 256) r[k] = lv[cnt++] * inv;
}

// out = LayerNorm(x + a) * g + b  (block per row; in-place safe)
__global__ __launch_bounds__(256)
void add_ln_kernel(const float* __restrict__ xin, const float* __restrict__ a,
                   const float* __restrict__ g, const float* __restrict__ bta,
                   float* __restrict__ out, int Dm) {
  __shared__ float sm[256];
  long long row = blockIdx.x;
  const float* xr = xin + row * Dm;
  const float* ar = a + row * Dm;
  float lv[8]; int cnt = 0; float s = 0.f;
  for (int d = threadIdx.x; d < Dm; d += 256) { float t = xr[d] + ar[d]; lv[cnt++] = t; s += t; }
  s = block_sum(s, sm);
  float mu = s / (float)Dm;
  float vs = 0.f;
  for (int i = 0; i < cnt; ++i) { float t = lv[i] - mu; vs += t * t; }
  vs = block_sum(vs, sm);
  float inv = rsqrtf(vs / (float)Dm + 1e-5f);
  cnt = 0;
  for (int d = threadIdx.x; d < Dm; d += 256)
    out[row * Dm + d] = (lv[cnt++] - mu) * inv * g[d] + bta[d];
}

// final softmax over V (streaming, block per row)
__global__ __launch_bounds__(256)
void vocab_softmax_kernel(const float* __restrict__ lg, float* __restrict__ out, int Vn) {
  __shared__ float sm[256];
  long long row = blockIdx.x;
  const float* r = lg + row * (long long)Vn;
  float m = -3.4e38f;
  for (int k = threadIdx.x; k < Vn; k += 256) m = fmaxf(m, r[k]);
  m = block_max(m, sm);
  float s = 0.f;
  for (int k = threadIdx.x; k < Vn; k += 256) s += __expf(r[k] - m);
  s = block_sum(s, sm);
  float inv = 1.f / s;
  for (int k = threadIdx.x; k < Vn; k += 256)
    out[row * (long long)Vn + k] = __expf(r[k] - m) * inv;
}

// ------------------------------ launcher -----------------------------------

extern "C" void kernel_launch(void* const* d_in, const int* in_sizes, int n_in,
                              void* d_out, int out_size, void* d_ws, size_t ws_size,
                              hipStream_t stream) {
  constexpr int Lc = 6, Dc = 1024, Hc = 16, DFFc = 4096, Vc = 32000;
  constexpr int Bc = 4, Sc = 512, DKc = 64;
  const int  Mtok = Bc * Sc;                 // 2048 tokens per stream
  const long long BH = (long long)Bc * Hc;   // 64 attention batches

  const int*   src     = (const int*)  d_in[0];
  const int*   tgt     = (const int*)  d_in[1];
  const float* enc_emb = (const float*)d_in[2];
  const float* dec_emb = (const float*)d_in[3];
  const float* eWq = (const float*)d_in[4],  *eWk = (const float*)d_in[5];
  const float* eWv = (const float*)d_in[6],  *eWo = (const float*)d_in[7];
  const float* eBq = (const float*)d_in[8],  *eBk = (const float*)d_in[9];
  const float* eBv = (const float*)d_in[10], *eBo = (const float*)d_in[11];
  const float* eL1g = (const float*)d_in[12], *eL1b = (const float*)d_in[13];
  const float* eL2g = (const float*)d_in[14], *eL2b = (const float*)d_in[15];
  const float* eF1  = (const float*)d_in[16], *eF1b = (const float*)d_in[17];
  const float* eF2  = (const float*)d_in[18], *eF2b = (const float*)d_in[19];
  const float* dsWq = (const float*)d_in[20], *dsWk = (const float*)d_in[21];
  const float* dsWv = (const float*)d_in[22], *dsWo = (const float*)d_in[23];
  const float* dcWq = (const float*)d_in[24], *dcWk = (const float*)d_in[25];
  const float* dcWv = (const float*)d_in[26], *dcWo = (const float*)d_in[27];
  const float* dsBq = (const float*)d_in[28], *dsBk = (const float*)d_in[29];
  const float* dsBv = (const float*)d_in[30], *dsBo = (const float*)d_in[31];
  const float* dcBq = (const float*)d_in[32], *dcBk = (const float*)d_in[33];
  const float* dcBv = (const float*)d_in[34], *dcBo = (const float*)d_in[35];
  const float* dL1g = (const float*)d_in[36], *dL2g = (const float*)d_in[37];
  const float* dL3g = (const float*)d_in[38];
  const float* dL1b = (const float*)d_in[39], *dL2b = (const float*)d_in[40];
  const float* dL3b = (const float*)d_in[41];
  const float* dF1  = (const float*)d_in[42], *dF1b = (const float*)d_in[43];
  const float* dF2  = (const float*)d_in[44], *dF2b = (const float*)d_in[45];
  const float* fW   = (const float*)d_in[46], *fb   = (const float*)d_in[47];

  // workspace arena (~463 MB of f32)
  float* ws = (float*)d_ws;
  size_t off = 0;
  auto arena = [&](size_t n) { float* p = ws + off; off += n; return p; };
  const size_t MD = (size_t)Mtok * Dc;
  float* x   = arena(MD);                        // encoder stream / memory
  float* y   = arena(MD);                        // decoder stream
  float* qb  = arena(MD);
  float* kb  = arena(MD);
  float* vb  = arena(MD);
  float* qh  = arena(MD);
  float* kh  = arena(MD);
  float* vh  = arena(MD);
  float* oh  = arena(MD);
  float* oc  = arena(MD);
  float* ab  = arena(MD);                        // attention output (post-Wo)
  float* ffb = arena((size_t)Mtok * DFFc);       // FFN hidden
  float* f2o = arena(MD);                        // FFN output
  float* scb = arena((size_t)BH * Sc * Sc);      // scores 64 x 512 x 512
  float* lgb = arena((size_t)Mtok * Vc);         // logits

  auto gemm = [&](const float* A, const float* Bm, const float* bias, float* Cc,
                  int M_, int N_, int K_, int transB_, int relu_,
                  long long sA_ = 0, long long sB_ = 0, long long sC_ = 0, int batch_ = 1) {
    dim3 g((unsigned)((N_ + TBN - 1) / TBN), (unsigned)((M_ + TBM - 1) / TBM), (unsigned)batch_);
    gemm_bf16_wmma<<<g, 256, 0, stream>>>(A, Bm, bias, Cc, M_, N_, K_, sA_, sB_, sC_, transB_, relu_);
  };
  auto rope = [&](float* p) {
    long long np = (long long)Mtok * Dc / 2;
    rope_kernel<<<(unsigned)((np + 255) / 256), 256, 0, stream>>>(p, np, Sc, Hc, DKc, Dc);
  };
  auto toH = [&](const float* a, float* b) {
    long long tot = (long long)Mtok * Dc;
    to_heads_kernel<<<(unsigned)((tot + 255) / 256), 256, 0, stream>>>(a, b, tot, Sc, Hc, DKc);
  };
  auto fromH = [&](const float* a, float* b) {
    long long tot = (long long)Mtok * Dc;
    from_heads_kernel<<<(unsigned)((tot + 255) / 256), 256, 0, stream>>>(a, b, tot, Sc, Hc, DKc);
  };
  auto addln = [&](float* io, const float* add, const float* g, const float* b) {
    add_ln_kernel<<<Mtok, 256, 0, stream>>>(io, add, g, b, io, Dc);
  };

  // One multi-head attention block: io = LN(io + MHA(qin, kvin))
  auto mha = [&](const float* qin, const float* kvin,
                 const float* Wq, const float* Bq, const float* Wk, const float* Bk,
                 const float* Wv, const float* Bv, const float* Wo, const float* Bo,
                 const int* toks, int causal,
                 float* io, const float* lg, const float* lb) {
    gemm(qin,  Wq, Bq, qb, Mtok, Dc, Dc, 0, 0);
    gemm(kvin, Wk, Bk, kb, Mtok, Dc, Dc, 0, 0);
    gemm(kvin, Wv, Bv, vb, Mtok, Dc, Dc, 0, 0);
    rope(qb); rope(kb);
    toH(qb, qh); toH(kb, kh); toH(vb, vh);
    // scores[z] = Qh[z] (SxDK) x Kh[z]^T  -> (SxS), batched over B*H
    gemm(qh, kh, nullptr, scb, Sc, Sc, DKc, /*transB=*/1, 0,
         (long long)Sc * DKc, (long long)Sc * DKc, (long long)Sc * Sc, (int)BH);
    attn_softmax_kernel<<<(unsigned)(BH * Sc), 256, 0, stream>>>(
        scb, toks, Sc, Sc, Hc, causal, 0.125f /* 1/sqrt(64) */);
    // out[z] = P[z] (SxS) x Vh[z] (SxDK)
    gemm(scb, vh, nullptr, oh, Sc, DKc, Sc, 0, 0,
         (long long)Sc * Sc, (long long)Sc * DKc, (long long)Sc * DKc, (int)BH);
    fromH(oh, oc);
    gemm(oc, Wo, Bo, ab, Mtok, Dc, Dc, 0, 0);
    addln(io, ab, lg, lb);
  };
  auto ffn = [&](float* io, const float* W1, const float* b1,
                 const float* W2, const float* b2, const float* g, const float* b) {
    gemm(io,  W1, b1, ffb, Mtok, DFFc, Dc, 0, /*relu=*/1);
    gemm(ffb, W2, b2, f2o, Mtok, Dc, DFFc, 0, 0);
    addln(io, f2o, g, b);
  };

  // ----- encoder -----
  embed_kernel<<<Mtok, 256, 0, stream>>>(src, enc_emb, x, Dc);
  for (int l = 0; l < Lc; ++l) {
    const long long wo = (long long)l * Dc * Dc, bo = (long long)l * Dc;
    mha(x, x, eWq + wo, eBq + bo, eWk + wo, eBk + bo, eWv + wo, eBv + bo,
        eWo + wo, eBo + bo, src, /*causal=*/0, x, eL1g + bo, eL1b + bo);
    ffn(x, eF1 + (long long)l * Dc * DFFc, eF1b + (long long)l * DFFc,
        eF2 + (long long)l * DFFc * Dc, eF2b + bo, eL2g + bo, eL2b + bo);
  }
  // x now holds encoder memory

  // ----- decoder -----
  embed_kernel<<<Mtok, 256, 0, stream>>>(tgt, dec_emb, y, Dc);
  for (int l = 0; l < Lc; ++l) {
    const long long wo = (long long)l * Dc * Dc, bo = (long long)l * Dc;
    // masked self-attention (pad + causal)
    mha(y, y, dsWq + wo, dsBq + bo, dsWk + wo, dsBk + bo, dsWv + wo, dsBv + bo,
        dsWo + wo, dsBo + bo, tgt, /*causal=*/1, y, dL1g + bo, dL1b + bo);
    // cross-attention (pad mask over source keys)
    mha(y, x, dcWq + wo, dcBq + bo, dcWk + wo, dcBk + bo, dcWv + wo, dcBv + bo,
        dcWo + wo, dcBo + bo, src, /*causal=*/0, y, dL2g + bo, dL2b + bo);
    ffn(y, dF1 + (long long)l * Dc * DFFc, dF1b + (long long)l * DFFc,
        dF2 + (long long)l * DFFc * Dc, dF2b + bo, dL3g + bo, dL3b + bo);
  }

  // ----- logits + softmax -----
  gemm(y, fW, fb, lgb, Mtok, Vc, Dc, 0, 0);
  vocab_softmax_kernel<<<Mtok, 256, 0, stream>>>(lgb, (float*)d_out, Vc);
}
